// MHSA_54056458387995
// MI455X (gfx1250) — compile-verified, matrix operand order checked by reading, not run on hardware
//
#include <hip/hip_runtime.h>
#include <hip/hip_bf16.h>

typedef __bf16 bf16_t;
typedef __attribute__((ext_vector_type(16))) __bf16 v16bf;
typedef __attribute__((ext_vector_type(8)))  __bf16 v8bf;
typedef __attribute__((ext_vector_type(4)))  __bf16 v4bf;
typedef __attribute__((ext_vector_type(8)))  float  v8f;
typedef __attribute__((ext_vector_type(4)))  unsigned v4u;
typedef __attribute__((ext_vector_type(8)))  int      v8i;
typedef __attribute__((ext_vector_type(4)))  int      v4i;

#define DIM      1024
#define HEADS    16
#define HEAD_DIM 64
#define SEQ      2048
#define BATCH    4

// ---------------------------------------------------------------------------
// Async global->LDS (GLOBAL_LOAD_ASYNC_TO_LDS_B128, ASYNCcnt-tracked).
// ---------------------------------------------------------------------------
#if __has_builtin(__builtin_amdgcn_global_load_async_to_lds_b128)
#define ASYNC_LDS 1
typedef int v4i_vs __attribute__((__vector_size__(4 * sizeof(int))));
typedef __attribute__((address_space(1))) v4i_vs* gptr_v4i;
typedef __attribute__((address_space(3))) v4i_vs* lptr_v4i;
__device__ __forceinline__ void async_copy16(const bf16_t* g, bf16_t* l) {
  __builtin_amdgcn_global_load_async_to_lds_b128(
      (gptr_v4i)(size_t)(g), (lptr_v4i)(unsigned)(size_t)(l), 0, 0);
}
__device__ __forceinline__ void async_wait0() {
#if __has_builtin(__builtin_amdgcn_s_wait_asynccnt)
  __builtin_amdgcn_s_wait_asynccnt(0);
#else
  __asm volatile("s_wait_asynccnt 0" ::: "memory");
#endif
}
#else
#define ASYNC_LDS 0
#endif

// ---------------------------------------------------------------------------
// Tensor Data Mover: one TENSOR_LOAD_TO_LDS per K-tile (CDNA5 ISA-8).
// D# group0: {count=1 | gather off, lds_addr, global_addr[31:0],
//             global_addr[56:32] | type=2<<30}
// D# group1: {wg_mask=0 | data_size=1(2B)<<16 | pad_enable<<20 |
//             pad_interval=4(32 DW)<<22 | pad_amount=3(4 DW)<<25,
//             tensor_dim0=64 (bits 79:48), tensor_dim1=2048 (bits 111:80),
//             tile_dim0=64 (bits 127:112), tile_dim1=32, tile_dim2=0,
//             tensor_dim0_stride=3072 (bits 207:160)}
// Pad fields recreate the LDS layout Ks[32][72] (32-DWORD rows + 4-DWORD pad).
// ---------------------------------------------------------------------------
#if __has_builtin(__builtin_amdgcn_tensor_load_to_lds)
#define TDM_OK 1
__device__ __forceinline__ void tdm_load_k_tile(const bf16_t* gsrc, bf16_t* lds) {
  const unsigned long long ga = (unsigned long long)(size_t)gsrc;
  v4u g0 = { 1u,
             (unsigned)(size_t)lds,
             (unsigned)ga,
             (unsigned)((ga >> 32) & 0x1FFFFFFu) | (2u << 30) };
  v8i g1 = { (int)((1u << 16) | (1u << 20) | (4u << 22) | (3u << 25)),
             (int)(64u << 16),        // tensor_dim0 = 64 (low16 @ bits 63:48)
             (int)(2048u << 16),      // tensor_dim1 = 2048 (low16 @ bits 95:80)
             (int)(64u << 16),        // tile_dim0 = 64 (bits 127:112)
             32,                      // tile_dim1 = 32, tile_dim2 = 0
             3 * DIM,                 // tensor_dim0_stride (elements)
             0, 0 };
  v4i gz = { 0, 0, 0, 0 };
#if defined(__clang_major__) && (__clang_major__ >= 23)
  v8i gz8 = { 0, 0, 0, 0, 0, 0, 0, 0 };
  __builtin_amdgcn_tensor_load_to_lds(g0, g1, gz, gz, gz8, 0);
#else
  __builtin_amdgcn_tensor_load_to_lds(g0, g1, gz, gz, 0);
#endif
}
__device__ __forceinline__ void tensor_wait0() {
#if __has_builtin(__builtin_amdgcn_s_wait_tensorcnt)
  __builtin_amdgcn_s_wait_tensorcnt(0);
#else
  __asm volatile("s_wait_tensorcnt 0" ::: "memory");
#endif
}
#else
#define TDM_OK 0
#endif

// ---------------------------------------------------------------------------
// 16-lane row reductions via DPP ROW_XMASK (pure VALU, co-executes with WMMA).
// ---------------------------------------------------------------------------
#if __has_builtin(__builtin_amdgcn_update_dpp)
template<int CTRL>
__device__ __forceinline__ float dpp_mov(float x) {
  return __builtin_bit_cast(float,
      __builtin_amdgcn_update_dpp(0, __builtin_bit_cast(int, x),
                                  CTRL, 0xF, 0xF, true));
}
__device__ __forceinline__ float red_max16(float v) {
  v = fmaxf(v, dpp_mov<0x168>(v));   // ROW_XMASK:8
  v = fmaxf(v, dpp_mov<0x164>(v));   // ROW_XMASK:4
  v = fmaxf(v, dpp_mov<0x162>(v));   // ROW_XMASK:2
  v = fmaxf(v, dpp_mov<0x161>(v));   // ROW_XMASK:1
  return v;
}
__device__ __forceinline__ float red_sum16(float v) {
  v += dpp_mov<0x168>(v);
  v += dpp_mov<0x164>(v);
  v += dpp_mov<0x162>(v);
  v += dpp_mov<0x161>(v);
  return v;
}
#else
__device__ __forceinline__ float red_max16(float v) {
  for (int off = 8; off >= 1; off >>= 1) v = fmaxf(v, __shfl_xor(v, off, 32));
  return v;
}
__device__ __forceinline__ float red_sum16(float v) {
  for (int off = 8; off >= 1; off >>= 1) v += __shfl_xor(v, off, 32);
  return v;
}
#endif

// ---------------------------------------------------------------------------
// Fragment helpers (CDNA5 wave32 WMMA layouts, CDNA5 ISA §7.12.2)
// ---------------------------------------------------------------------------
__device__ __forceinline__ v16bf load_frag_k32(const bf16_t* __restrict__ base,
                                               int stride, int lane) {
  const int r  = lane & 15;
  const int k0 = (lane >> 4) << 3;
  const bf16_t* p = base + r * stride + k0;
  v8bf lo = *(const v8bf*)(p);
  v8bf hi = *(const v8bf*)(p + 16);
  return __builtin_shufflevector(lo, hi, 0,1,2,3,4,5,6,7,8,9,10,11,12,13,14,15);
}

__device__ __forceinline__ v8f wmma_bf16(v16bf a, v16bf b, v8f c) {
  return __builtin_amdgcn_wmma_f32_16x16x32_bf16(false, a, false, b,
                                                 (short)0, c, false, false);
}

// ---------------------------------------------------------------------------
// f32 -> bf16 conversion (vectorized x4)
// ---------------------------------------------------------------------------
__global__ void f32_to_bf16_kernel(const float4* __restrict__ src,
                                   v4bf* __restrict__ dst, int n4) {
  int i = blockIdx.x * blockDim.x + threadIdx.x;
  int stride = gridDim.x * blockDim.x;
  for (; i < n4; i += stride) {
    float4 v = src[i];
    v4bf o;
    o[0] = (bf16_t)v.x; o[1] = (bf16_t)v.y;
    o[2] = (bf16_t)v.z; o[3] = (bf16_t)v.w;
    dst[i] = o;
  }
}

// ---------------------------------------------------------------------------
// Tiled bf16 WMMA GEMM with bias: C[M,N] = A[M,K] @ B[K,N] + bias
// Block tile 128x128, 256 threads = 8 waves, each wave 32x64 (2x4 WMMA frags).
// ---------------------------------------------------------------------------
template<bool OUT_BF16>
__global__ __launch_bounds__(256) void gemm_bias_kernel(
    const bf16_t* __restrict__ A, const bf16_t* __restrict__ B,
    const float* __restrict__ bias, void* __restrict__ Cout,
    int M, int N, int K)
{
  __shared__ alignas(16) bf16_t As[128][40];   // [m][k], pad to 40 (80B rows)
  __shared__ alignas(16) bf16_t Bt[128][40];   // [n][k] (transposed)

  const int tid  = threadIdx.x;
  const int lane = tid & 31;
  const int w    = tid >> 5;
  const int wm   = (w >> 1) * 32;
  const int wn   = (w & 1) * 64;
  const int rowBase = blockIdx.y * 128;
  const int colBase = blockIdx.x * 128;

  v8f acc[2][4] = {};

  const int ar = tid >> 1;            // 0..127 (A row)
  const int ak = (tid & 1) * 16;      // 0/16   (A k half)
  const int bk = tid >> 3;            // 0..31  (B k row)
  const int bn = (tid & 7) * 16;      // 0..112 (B col chunk)

  for (int kb = 0; kb < K; kb += 32) {
    {   // A tile: 16 contiguous bf16 per thread (two async B128 transfers)
      const bf16_t* src = A + (size_t)(rowBase + ar) * K + kb + ak;
#if ASYNC_LDS
      async_copy16(src,     &As[ar][ak]);
      async_copy16(src + 8, &As[ar][ak + 8]);
#else
      *(v8bf*)&As[ar][ak]     = *(const v8bf*)(src);
      *(v8bf*)&As[ar][ak + 8] = *(const v8bf*)(src + 8);
#endif
    }
    {   // B tile: read row-major, scatter transposed into Bt[n][k]
      const bf16_t* src = B + (size_t)(kb + bk) * N + colBase + bn;
      bf16_t tmp[16];
      *(v8bf*)(tmp)     = *(const v8bf*)(src);
      *(v8bf*)(tmp + 8) = *(const v8bf*)(src + 8);
      #pragma unroll
      for (int i = 0; i < 16; ++i) Bt[bn + i][bk] = tmp[i];
    }
    if (kb + 32 < K)
      __builtin_prefetch(A + (size_t)(rowBase + ar) * K + kb + 32 + ak, 0, 1);
#if ASYNC_LDS
    async_wait0();
#endif
    __syncthreads();

    v16bf af[2], bfr[4];
    #pragma unroll
    for (int mi = 0; mi < 2; ++mi)
      af[mi] = load_frag_k32(&As[wm + mi * 16][0], 40, lane);
    #pragma unroll
    for (int ni = 0; ni < 4; ++ni)
      bfr[ni] = load_frag_k32(&Bt[wn + ni * 16][0], 40, lane);
    #pragma unroll
    for (int mi = 0; mi < 2; ++mi)
      #pragma unroll
      for (int ni = 0; ni < 4; ++ni)
        acc[mi][ni] = wmma_bf16(af[mi], bfr[ni], acc[mi][ni]);
    __syncthreads();
  }

  const int hi8 = (lane >> 4) * 8;
  #pragma unroll
  for (int mi = 0; mi < 2; ++mi) {
    #pragma unroll
    for (int ni = 0; ni < 4; ++ni) {
      const int col = colBase + wn + ni * 16 + (lane & 15);
      const float bv = bias[col];
      #pragma unroll
      for (int j = 0; j < 8; ++j) {
        const int row = rowBase + wm + mi * 16 + hi8 + j;
        const float v = acc[mi][ni][j] + bv;
        if (OUT_BF16) ((bf16_t*)Cout)[(size_t)row * N + col] = (bf16_t)v;
        else          ((float*)Cout)[(size_t)row * N + col]  = v;
      }
    }
  }
}

// ---------------------------------------------------------------------------
// Flash attention: grid (S/64, H, B), block 128 = 4 waves, wave = 16 q-rows.
// K tile moved by the Tensor Data Mover (one TENSOR_LOAD_TO_LDS per tile,
// issued by wave 0, TENSORcnt-synced); V tile transposed into LDS; online
// softmax with DPP ROW_XMASK reductions.
// ---------------------------------------------------------------------------
__global__ __launch_bounds__(128) void attn_kernel(
    const bf16_t* __restrict__ qkv, bf16_t* __restrict__ ctx)
{
  __shared__ alignas(16) bf16_t Ks[32][72];     // [key][d] (TDM pad layout)
  __shared__ alignas(16) bf16_t Vt[64][40];     // [d][key]  (transposed)
  __shared__ alignas(16) bf16_t Ps[4][16][40];  // per-wave P, [row][key]

  const int tid  = threadIdx.x;
  const int lane = tid & 31;
  const int w    = tid >> 5;
  const int qb   = blockIdx.x * 64;
  const int h    = blockIdx.y;
  const int b    = blockIdx.z;
  const size_t tok0 = (size_t)b * SEQ;
  const float scale = 0.125f;                   // 1/sqrt(64)

  // Q fragments (A layout), loaded once: two d-chunks of 32
  const int r  = lane & 15;
  const int k0 = (lane >> 4) << 3;
  const int qrow = qb + w * 16 + r;
  const bf16_t* qp = qkv + (tok0 + qrow) * (3 * DIM) + h * HEAD_DIM;
  v16bf qa[2];
  #pragma unroll
  for (int c = 0; c < 2; ++c) {
    v8bf lo = *(const v8bf*)(qp + c * 32 + k0);
    v8bf hi = *(const v8bf*)(qp + c * 32 + k0 + 16);
    qa[c] = __builtin_shufflevector(lo, hi, 0,1,2,3,4,5,6,7,8,9,10,11,12,13,14,15);
  }

  float m[8], l[8];
  v8f o[4] = {};
  #pragma unroll
  for (int j = 0; j < 8; ++j) { m[j] = -3.0e38f; l[j] = 0.f; }

  const int lrow = tid >> 2;        // 0..31 : key within tile
  const int ld0  = (tid & 3) * 16;  // 0..48 : d chunk

  for (int kt = 0; kt < SEQ; kt += 32) {
    {   // K tile -> LDS [key][d] via TDM (fallback: async per-lane copies)
      const bf16_t* ksrc = qkv + (tok0 + kt) * (3 * DIM) + DIM + h * HEAD_DIM;
#if TDM_OK
      if (w == 0) {
        tdm_load_k_tile(ksrc, &Ks[0][0]);
        tensor_wait0();
      }
#elif ASYNC_LDS
      const bf16_t* src = ksrc + (size_t)lrow * (3 * DIM) + ld0;
      async_copy16(src,     &Ks[lrow][ld0]);
      async_copy16(src + 8, &Ks[lrow][ld0 + 8]);
#else
      const bf16_t* src = ksrc + (size_t)lrow * (3 * DIM) + ld0;
      *(v8bf*)&Ks[lrow][ld0]     = *(const v8bf*)(src);
      *(v8bf*)&Ks[lrow][ld0 + 8] = *(const v8bf*)(src + 8);
#endif
    }
    {   // V tile -> LDS transposed [d][key]
      const bf16_t* src = qkv + (tok0 + kt + lrow) * (3 * DIM) + 2 * DIM + h * HEAD_DIM + ld0;
      bf16_t tmp[16];
      *(v8bf*)(tmp)     = *(const v8bf*)(src);
      *(v8bf*)(tmp + 8) = *(const v8bf*)(src + 8);
      #pragma unroll
      for (int i = 0; i < 16; ++i) Vt[ld0 + i][lrow] = tmp[i];
    }
    if (kt + 32 < SEQ)   // prefetch next V tile (global_prefetch_b8)
      __builtin_prefetch(qkv + (tok0 + kt + 32 + lrow) * (3 * DIM) + 2 * DIM + h * HEAD_DIM + ld0, 0, 1);
#if !TDM_OK && ASYNC_LDS
    async_wait0();
#endif
    __syncthreads();

    // S = scale * Q @ K^T  (two 16-col subtiles, K-dim = 64 in two chunks)
    v8f s0 = {}, s1 = {};
    s0 = wmma_bf16(qa[0], load_frag_k32(&Ks[0][0],   72, lane), s0);
    s0 = wmma_bf16(qa[1], load_frag_k32(&Ks[0][32],  72, lane), s0);
    s1 = wmma_bf16(qa[0], load_frag_k32(&Ks[16][0],  72, lane), s1);
    s1 = wmma_bf16(qa[1], load_frag_k32(&Ks[16][32], 72, lane), s1);
    s0 *= scale; s1 *= scale;

    // online softmax update, per accumulator row (VGPR index j)
    #pragma unroll
    for (int j = 0; j < 8; ++j) {
      const float tmax  = red_max16(fmaxf(s0[j], s1[j]));
      const float mn    = fmaxf(m[j], tmax);
      const float alpha = __expf(m[j] - mn);
      const float p0 = __expf(s0[j] - mn);
      const float p1 = __expf(s1[j] - mn);
      const float ts = red_sum16(p0 + p1);
      l[j] = l[j] * alpha + ts;
      m[j] = mn;
      #pragma unroll
      for (int ni = 0; ni < 4; ++ni) o[ni][j] *= alpha;
      // redistribute P from C-layout to A-layout via per-wave LDS
      Ps[w][(lane >> 4) * 8 + j][lane & 15]        = (bf16_t)p0;
      Ps[w][(lane >> 4) * 8 + j][16 + (lane & 15)] = (bf16_t)p1;
    }
    __asm volatile("s_wait_dscnt 0" ::: "memory");  // wave-local LDS RAW

    // O += P @ V   (K-dim = 32 keys, N = 64 d-cols in four subtiles)
    v16bf pa = load_frag_k32(&Ps[w][0][0], 40, lane);
    #pragma unroll
    for (int ni = 0; ni < 4; ++ni)
      o[ni] = wmma_bf16(pa, load_frag_k32(&Vt[ni * 16][0], 40, lane), o[ni]);

    __syncthreads();
  }

  // epilogue: O /= l, store bf16 context [token][h*64+d]
  #pragma unroll
  for (int ni = 0; ni < 4; ++ni) {
    const int d = ni * 16 + (lane & 15);
    #pragma unroll
    for (int j = 0; j < 8; ++j) {
      const int row = qb + w * 16 + (lane >> 4) * 8 + j;
      const float v = o[ni][j] / l[j];
      ctx[(tok0 + row) * (size_t)DIM + h * HEAD_DIM + d] = (bf16_t)v;
    }
  }
}

// ---------------------------------------------------------------------------
// Host-side orchestration
// ---------------------------------------------------------------------------
extern "C" void kernel_launch(void* const* d_in, const int* in_sizes, int n_in,
                              void* d_out, int out_size, void* d_ws, size_t ws_size,
                              hipStream_t stream) {
  const float* x      = (const float*)d_in[0];   // [4,2048,1024]
  const float* W_qkv  = (const float*)d_in[1];   // [1024,3072]
  const float* b_qkv  = (const float*)d_in[2];   // [3072]
  const float* W_proj = (const float*)d_in[3];   // [1024,1024]
  const float* b_proj = (const float*)d_in[4];   // [1024]
  float* out = (float*)d_out;                    // [4,2048,1024] f32

  char* ws = (char*)d_ws;
  bf16_t* xb  = (bf16_t*)(ws);                                  // 16 MB
  bf16_t* wqb = (bf16_t*)(ws + 16777216);                       //  6 MB
  bf16_t* wpb = (bf16_t*)(ws + 16777216 + 6291456);             //  2 MB
  bf16_t* qkv = (bf16_t*)(ws + 16777216 + 6291456 + 2097152);   // 48 MB
  bf16_t* ctx = (bf16_t*)(ws + 16777216 + 6291456 + 2097152 + 50331648); // 16 MB

  // fp32 -> bf16 staging
  f32_to_bf16_kernel<<<4096, 256, 0, stream>>>((const float4*)x,      (v4bf*)xb,  (BATCH*SEQ*DIM)/4);
  f32_to_bf16_kernel<<<3072, 256, 0, stream>>>((const float4*)W_qkv,  (v4bf*)wqb, (DIM*3*DIM)/4);
  f32_to_bf16_kernel<<<1024, 256, 0, stream>>>((const float4*)W_proj, (v4bf*)wpb, (DIM*DIM)/4);

  // qkv = x @ W_qkv + b_qkv  -> bf16 scratch [8192, 3072]
  gemm_bias_kernel<true><<<dim3(3*DIM/128, BATCH*SEQ/128), 256, 0, stream>>>(
      xb, wqb, b_qkv, qkv, BATCH*SEQ, 3*DIM, DIM);

  // flash attention per (q-block, head, batch) -> bf16 ctx [8192, 1024]
  attn_kernel<<<dim3(SEQ/64, HEADS, BATCH), 128, 0, stream>>>(qkv, ctx);

  // out = ctx @ W_proj + b_proj -> f32 d_out
  gemm_bias_kernel<false><<<dim3(DIM/128, BATCH*SEQ/128), 256, 0, stream>>>(
      ctx, wpb, b_proj, out, BATCH*SEQ, DIM, DIM);
}
